// minGRU_58334245814925
// MI455X (gfx1250) — matrix operand order repaired and checked by reading, not compiled, not run on hardware
//
#include <hip/hip_runtime.h>
#include <hip/hip_bf16.h>

typedef __attribute__((ext_vector_type(16))) __bf16 v16bf;
typedef __attribute__((ext_vector_type(8)))  __bf16 v8bf;
typedef __attribute__((ext_vector_type(4)))  __bf16 v4bf;
typedef __attribute__((ext_vector_type(8)))  float  v8f;

#define B_     4
#define S_     4096
#define D_     1024
#define K_     1024
#define NC_    32
#define CHUNK_ 128

// Branchless fast sigmoid: v_exp_f32 + v_rcp_f32, no IEEE divide, no branches.
__device__ __forceinline__ float fast_sigmoid(float x) {
    return __builtin_amdgcn_rcpf(1.0f + __expf(-x));
}

// ---------------------------------------------------------------------------
// fp32 -> bf16 pre-conversion (one pass, bandwidth bound, L2-resident after).
// ---------------------------------------------------------------------------
__global__ void __launch_bounds__(256)
cvt_f32_bf16_kernel(const float4* __restrict__ src, v4bf* __restrict__ dst, int n4)
{
    const int i = blockIdx.x * blockDim.x + threadIdx.x;
    if (i < n4) {
        const float4 f = src[i];
        v4bf o;
        o[0] = (__bf16)f.x; o[1] = (__bf16)f.y;
        o[2] = (__bf16)f.z; o[3] = (__bf16)f.w;
        dst[i] = o;
    }
}

// ---------------------------------------------------------------------------
// GEMM + fused activation. Block = 4 waves sharing one 16-row M-tile.
// The shared A tile (16x32 bf16 = 1KB per K-step) is staged into LDS with
// global_load_async_to_lds_b128 (ASYNCcnt) by waves 0-1; each wave then owns
// one N-tile pair (hidden col n, gate col D+n) -> 2 WMMAs per K-step with a
// shared A fragment read from LDS. Epilogue computes coeff/value in registers.
// ---------------------------------------------------------------------------
__global__ void __launch_bounds__(128)
gemm_act_kernel(const __bf16* __restrict__ xb, const __bf16* __restrict__ Wb,
                const long long* __restrict__ ids,
                float* __restrict__ coeff, float* __restrict__ value)
{
    __shared__ __attribute__((aligned(32))) __bf16 Atile[16 * 32];

    const int tid    = threadIdx.x;
    const int lane   = tid & 31;
    const int wave   = tid >> 5;
    const int mTile  = blockIdx.x >> 4;          // 1024 M-tiles
    const int nTile  = (blockIdx.x & 15) * 4 + wave;
    const int m0     = mTile * 16;
    const int n0     = nTile * 16;

    // A-frag addressing (16x32 bf16): row = lane&15; 16 consecutive K values
    // {akb..akb+15} with akb = 0 (lanes 0-15) or 16 (lanes 16-31).
    const int arow = lane & 15;
    const int akb  = (lane < 16) ? 0 : 16;
    // B-frag addressing (32x16 bf16 KxN): col = n + (lane&15), 16 consecutive
    // K values starting at bkb = 0 (lanes 0-15) or 16 (lanes 16-31).
    const int bkb  = (lane < 16) ? 0 : 16;
    const __bf16* wrow_h = Wb + (size_t)(n0 + (lane & 15)) * K_;
    const __bf16* wrow_g = Wb + (size_t)(D_ + n0 + (lane & 15)) * K_;

    // Async staging assignment: threads 0-63, one b128 (8 bf16) each:
    // row = tid>>2, 16-byte segment = tid&3.
    const int srow = tid >> 2;
    const int sseg = tid & 3;
    const unsigned ldsDst =
        (unsigned)(uintptr_t)(&Atile[srow * 32]) + (unsigned)(sseg * 16);
    const __bf16* gsrc = xb + (size_t)(m0 + srow) * K_ + sseg * 8;

    v8f acc_h = {};
    v8f acc_g = {};

    for (int k0 = 0; k0 < K_; k0 += 32) {
        if (tid < 64) {                          // wave-uniform (waves 0,1)
            const unsigned long long ga = (unsigned long long)(uintptr_t)(gsrc + k0);
            asm volatile("global_load_async_to_lds_b128 %0, %1, off"
                         :: "v"(ldsDst), "v"(ga) : "memory");
            asm volatile("s_wait_asynccnt 0x0" ::: "memory");
        }
        __syncthreads();                         // A tile visible to all waves

        // Unconditional speculative prefetch of next K-step of W rows.
        __builtin_prefetch(wrow_h + k0 + 32, 0, 3);
        __builtin_prefetch(wrow_g + k0 + 32, 0, 3);

        // A fragment from LDS: two 16B ds reads, concatenated.
        const v8bf a_lo = *(const v8bf*)(&Atile[arow * 32 + akb]);
        const v8bf a_hi = *(const v8bf*)(&Atile[arow * 32 + 8 + akb]);
        const v16bf a = __builtin_shufflevector(a_lo, a_hi,
                            0, 1, 2, 3, 4, 5, 6, 7, 8, 9, 10, 11, 12, 13, 14, 15);

        // B fragments: 32B contiguous (32B-aligned) global loads.
        const v16bf bh = *(const v16bf*)(wrow_h + k0 + bkb);
        const v16bf bg = *(const v16bf*)(wrow_g + k0 + bkb);

        acc_h = __builtin_amdgcn_wmma_f32_16x16x32_bf16(
                    false, a, false, bh, (short)0, acc_h, false, false);
        acc_g = __builtin_amdgcn_wmma_f32_16x16x32_bf16(
                    false, a, false, bg, (short)0, acc_g, false, false);

        __syncthreads();                         // protect Atile before restage
    }

    // Epilogue (branchless): C/D layout — VGPR r holds row m0+r+(lane<16?0:8),
    // column n0 + (lane&15).
    const int roff = (lane < 16) ? 0 : 8;
    const int n    = n0 + (lane & 15);
    #pragma unroll
    for (int r = 0; r < 8; ++r) {
        const int   m   = m0 + r + roff;
        const int   s   = m & (S_ - 1);
        const int   b   = m >> 12;               // m / S_
        const float hid = acc_h[r];
        const float gat = acc_g[r];

        const float z  = fast_sigmoid(gat);                   // sigmoid(gate)
        const float sh = fast_sigmoid(hid);
        const float g  = (hid >= 0.0f) ? (hid + 0.5f) : sh;   // v_cndmask
        float c = 1.0f - z;                                   // exp(-softplus(gate))
        float v = z * g;                                      // exp(log_z + log_g)

        const bool  rst  = (s > 0) && (ids[(size_t)b * S_ + (s - 1)] == 0);
        const float keep = rst ? 0.0f : 1.0f;                 // exp(PENALTY) ~ 0
        c *= keep;
        v *= keep;

        const size_t o = (size_t)m * D_ + n;
        coeff[o] = c;
        value[o] = v;
    }
}

// ---------------------------------------------------------------------------
// Chunked scan, pass A: per (b, chunk, d) compute A = prod(coeff) and
// B = local recurrence result starting from h = 0.
// ---------------------------------------------------------------------------
__global__ void __launch_bounds__(256)
scan_chunk_stats(const float* __restrict__ coeff, const float* __restrict__ value,
                 float* __restrict__ chA, float* __restrict__ chB)
{
    const int gid = blockIdx.x * blockDim.x + threadIdx.x;   // B*NC*D
    const int d = gid % D_;
    const int c = (gid / D_) % NC_;
    const int b = gid / (D_ * NC_);

    size_t base = ((size_t)b * S_ + (size_t)c * CHUNK_) * D_ + d;
    float A = 1.0f, h = 0.0f;
    for (int s = 0; s < CHUNK_; ++s) {
        const float cc = coeff[base];
        const float vv = value[base];
        h = fmaf(cc, h, vv);
        A *= cc;
        base += D_;
    }
    const size_t o = ((size_t)b * NC_ + c) * D_ + d;
    chA[o] = A;
    chB[o] = h;
}

// ---------------------------------------------------------------------------
// Pass B: serial combine of 32 chunk summaries per channel -> carry-in h.
// ---------------------------------------------------------------------------
__global__ void __launch_bounds__(256)
scan_carry(const float* __restrict__ chA, const float* __restrict__ chB,
           float* __restrict__ carry)
{
    const int gid = blockIdx.x * blockDim.x + threadIdx.x;   // B*D
    const int d = gid % D_;
    const int b = gid / D_;
    float h = 0.0f;
    for (int c = 0; c < NC_; ++c) {
        const size_t o = ((size_t)b * NC_ + c) * D_ + d;
        carry[o] = h;
        h = fmaf(chA[o], h, chB[o]);
    }
}

// ---------------------------------------------------------------------------
// Pass C: re-run recurrence per chunk from its carry-in, writing out.
// Last chunk also emits next_prev_hidden (zeroed where last token is pad).
// ---------------------------------------------------------------------------
__global__ void __launch_bounds__(256)
scan_apply(const float* __restrict__ coeff, const float* __restrict__ value,
           const float* __restrict__ carry, const long long* __restrict__ ids,
           float* __restrict__ out, float* __restrict__ nexth)
{
    const int gid = blockIdx.x * blockDim.x + threadIdx.x;   // B*NC*D
    const int d = gid % D_;
    const int c = (gid / D_) % NC_;
    const int b = gid / (D_ * NC_);

    float h = carry[((size_t)b * NC_ + c) * D_ + d];
    size_t base = ((size_t)b * S_ + (size_t)c * CHUNK_) * D_ + d;
    for (int s = 0; s < CHUNK_; ++s) {
        h = fmaf(coeff[base], h, value[base]);
        out[base] = h;
        base += D_;
    }
    if (c == NC_ - 1) {
        const bool pad = (ids[(size_t)b * S_ + (S_ - 1)] == 0);
        nexth[(size_t)b * D_ + d] = pad ? 0.0f : h;
    }
}

extern "C" void kernel_launch(void* const* d_in, const int* in_sizes, int n_in,
                              void* d_out, int out_size, void* d_ws, size_t ws_size,
                              hipStream_t stream)
{
    const float*     x   = (const float*)d_in[0];       // [B,S,D] fp32
    const float*     W   = (const float*)d_in[1];       // [2D,D]  fp32
    const long long* ids = (const long long*)d_in[2];   // [B,S]   int64

    float* out   = (float*)d_out;                       // [B,S,D]
    float* nexth = out + (size_t)B_ * S_ * D_;          // [B,1,D]

    const size_t BSD = (size_t)B_ * S_ * D_;            // 16,777,216
    const size_t BND = (size_t)B_ * NC_ * D_;           // 131,072
    const size_t WSZ = (size_t)2 * D_ * K_;             // 2,097,152

    // Workspace: coeff | value | chA | chB | carry | xb(bf16) | Wb(bf16)
    float*  coeff = (float*)d_ws;
    float*  value = coeff + BSD;
    float*  chA   = value + BSD;
    float*  chB   = chA + BND;
    float*  carry = chB + BND;
    __bf16* xbuf  = (__bf16*)(carry + BND);
    __bf16* Wbuf  = xbuf + BSD;

    // Pre-convert x and W to bf16 (vectorized float4 -> 4x bf16).
    {
        const int n4x = (int)(BSD / 4);
        cvt_f32_bf16_kernel<<<n4x / 256, 256, 0, stream>>>(
            (const float4*)x, (v4bf*)xbuf, n4x);
        const int n4w = (int)(WSZ / 4);
        cvt_f32_bf16_kernel<<<(n4w + 255) / 256, 256, 0, stream>>>(
            (const float4*)W, (v4bf*)Wbuf, n4w);
    }

    // GEMM: 1024 M-tiles x 16 N-groups (4 waves/blk, one N-tile pair each).
    gemm_act_kernel<<<1024 * 16, 128, 0, stream>>>(xbuf, Wbuf, ids, coeff, value);

    const int tChunks = B_ * NC_ * D_;                  // 131072
    scan_chunk_stats<<<tChunks / 256, 256, 0, stream>>>(coeff, value, chA, chB);

    const int tChan = B_ * D_;                          // 4096
    scan_carry<<<tChan / 256, 256, 0, stream>>>(chA, chB, carry);

    scan_apply<<<tChunks / 256, 256, 0, stream>>>(coeff, value, carry, ids,
                                                  out, nexth);
}
